// Message_gcn_21526376088371
// MI455X (gfx1250) — matrix operand order repaired and checked by reading, not compiled, not run on hardware
//
#include <hip/hip_runtime.h>

// ---------------- problem constants ----------------
constexpr int kBS = 32;
constexpr int kN  = 256;
constexpr int kE  = 64;
constexpr int kC  = 512;
constexpr int kH  = 4;
constexpr int kL  = 2;
constexpr int kM  = kE + 1;   // 65 hyperedges (0 = global)

#define CDIV(a, b) (((a) + (b) - 1) / (b))

typedef __attribute__((ext_vector_type(16))) _Float16 v16h;
typedef __attribute__((ext_vector_type(8)))  _Float16 v8h;
typedef __attribute__((ext_vector_type(4)))  _Float16 v4h;
typedef __attribute__((ext_vector_type(8)))  float    v8f;
typedef __attribute__((ext_vector_type(4)))  unsigned uv4;
typedef __attribute__((ext_vector_type(4)))  int      iv4;
typedef __attribute__((ext_vector_type(8)))  int      iv8;

// Use the Tensor Data Mover for B-tile staging when the builtin exists.
#if defined(__has_builtin)
#if __has_builtin(__builtin_amdgcn_tensor_load_to_lds) && __has_builtin(__builtin_amdgcn_s_wait_tensorcnt)
#define USE_TDM 1
#endif
#endif
#ifndef USE_TDM
#define USE_TDM 0
#endif

// LDS A-tile row stride in halves. 72 halves = 144 B: multiple of 16 B
// (aligned ds_load_b128) and row*36 dwords mod 64 banks distinct over 16 rows.
constexpr int kLdsStride = 72;

#if USE_TDM
// Issue one TDM 2D-tile load: 128 (x) by 64 (y) f32 elements, row stride ldb,
// from gptr into LDS byte-address ldsAddr.  dim0/dim1 bound the valid region
// from the tile start (hardware zero-fills OOB reads).
__device__ __forceinline__ void tdm_load_b(const float* gptr, unsigned ldsAddr,
                                           unsigned dim0, unsigned dim1, int ldb)
{
    unsigned long long ga  = (unsigned long long)(const void*)gptr;
    unsigned long long st0 = (unsigned long long)(unsigned)ldb;
    uv4 g0;
    g0[0] = 1u;                                                  // count=1 (valid D#)
    g0[1] = ldsAddr;                                             // lds_addr (bytes)
    g0[2] = (unsigned)(ga & 0xFFFFFFFFull);                      // global_addr lo
    g0[3] = (unsigned)((ga >> 32) & 0x01FFFFFFull) | (2u << 30); // addr hi | type=2
    iv8 g1;
    g1[0] = (int)(2u << 16);                                     // data_size = 4 bytes
    g1[1] = (int)((dim0 & 0xFFFFu) << 16);                       // tensor_dim0[15:0]
    g1[2] = (int)((dim0 >> 16) | ((dim1 & 0xFFFFu) << 16));      // dim0 hi | dim1 lo
    g1[3] = (int)((dim1 >> 16) | (128u << 16));                  // dim1 hi | tile_dim0=128
    g1[4] = (int)64u;                                            // tile_dim1=64, tile_dim2=0
    g1[5] = (int)(st0 & 0xFFFFFFFFull);                          // tensor_dim0_stride lo
    g1[6] = (int)((st0 >> 32) & 0xFFFFull);                      // stride hi | dim1_stride=0
    g1[7] = 0;
    iv4 g2 = {0, 0, 0, 0};
    iv4 g3 = {0, 0, 0, 0};
#if __clang_major__ >= 23
    iv8 g4 = {0, 0, 0, 0, 0, 0, 0, 0};
    __builtin_amdgcn_tensor_load_to_lds(g0, g1, g2, g3, g4, 0);
#else
    __builtin_amdgcn_tensor_load_to_lds(g0, g1, g2, g3, 0);
#endif
}
#endif

// =====================================================================
// Generic batched GEMM:  C[b] = A[b] (MxK) * B[b] (KxN), f32 in/out,
// f16 WMMA compute with f32 accumulation.
//  - block: 256 threads / 8 waves -> one 16x128 C tile, K-step 64
//  - A tile (16x64) staged sync to LDS as f16
//  - B tile (64x128) staged via TDM tensor_load_to_lds, DOUBLE-BUFFERED:
//    DMA of tile i+1 overlaps compute of tile i; K-tail zero-filled by HW
//  - fallback (no TDM builtin): sync transposed f16 staging
//  - requires Ncols % 128 == 0 (all call sites satisfy this)
// =====================================================================
__global__ __launch_bounds__(256)
void wmma_gemm_f16(const float* __restrict__ A, const float* __restrict__ B,
                   float* __restrict__ Cmat,
                   int M, int Ncols, int K,
                   int lda, int ldb, int ldc,
                   long long sA, long long sB, long long sC)
{
    __shared__ _Float16 As[16 * kLdsStride];    // [row][k] f16
#if USE_TDM
    __shared__ float Bs32[2][64 * 128];         // [k][n] f32, TDM double buffer
#else
    __shared__ _Float16 Bs[128 * kLdsStride];   // [col][k] f16 (transposed tile)
#endif

    const int b = blockIdx.z;
    A    += (long long)b * sA;
    B    += (long long)b * sB;
    Cmat += (long long)b * sC;

    const int tid  = threadIdx.x;
    const int lane = tid & 31;
    const int wave = tid >> 5;
    const int m0 = blockIdx.y * 16;
    const int n0 = blockIdx.x * 128;

    // ---- A staging roles (16x64 tile, 4 elems/thread) ----
    const int   sa_row = tid >> 4;
    const int   sa_k   = (tid & 15) * 4;
    const int   arow   = m0 + sa_row;
    const float amask  = (arow < M) ? 1.f : 0.f;
    const float* ArowCl = A + (long long)((arow < M) ? arow : (M - 1)) * lda;
    const float* aPtr   = ArowCl + sa_k;

#if !USE_TDM
    // ---- B staging roles: column tid&127, k-range (tid>>7)*32 .. +32 ----
    const int sb_col = tid & 127;
    const int sb_k0  = (tid >> 7) * 32;
    const float* Bcol = B + (n0 + sb_col);
    const float* bPtr = Bcol + (long long)sb_k0 * ldb;
    const long long bStep = (long long)64 * ldb;
#endif

    // ---- fragment roles (contiguous k = 16*(lane>>4) + s, same order in A and B) ----
    const int fa_row = lane & 15;
    const int fa_k   = (lane >> 4) * 16;       // A: K half 0..15 / 16..31
    const int fb_col = wave * 16 + (lane & 15);
    const int fb_k   = (lane >> 4) * 16;       // B: K half 0..15 / 16..31

#if USE_TDM
    const float* bsRead  = &Bs32[0][0];
    const float* bsWrite = &Bs32[1][0];
    unsigned ldsRead  = (unsigned)(unsigned long long)(const void*)bsRead;
    unsigned ldsWrite = (unsigned)(unsigned long long)(const void*)bsWrite;
    const float* Btile = B + n0;

    // prologue: start DMA of the first B tile into buffer 0
    if (wave == 0)
        tdm_load_b(Btile, ldsRead, (unsigned)(Ncols - n0), (unsigned)K, ldb);
#endif

    v8f acc = {0.f, 0.f, 0.f, 0.f, 0.f, 0.f, 0.f, 0.f};

    for (int k0 = 0; k0 < K; k0 += 64) {
        __syncthreads();   // previous iteration's LDS readers are done

        // ---------- stage A (sync, f16) ----------
        if (k0 + 64 <= K) {
            float a0 = aPtr[0], a1 = aPtr[1], a2 = aPtr[2], a3 = aPtr[3];
            v4h ah;
            ah[0] = (_Float16)(a0 * amask);
            ah[1] = (_Float16)(a1 * amask);
            ah[2] = (_Float16)(a2 * amask);
            ah[3] = (_Float16)(a3 * amask);
            *(v4h*)&As[sa_row * kLdsStride + sa_k] = ah;
        } else {
            v4h ah;
#pragma unroll
            for (int i = 0; i < 4; ++i) {
                int kk = k0 + sa_k + i;
                int kc = (kk < K) ? kk : (K - 1);
                float mk = (kk < K) ? amask : 0.f;
                ah[i] = (_Float16)(ArowCl[kc] * mk);
            }
            *(v4h*)&As[sa_row * kLdsStride + sa_k] = ah;
        }
        aPtr += 64;

#if USE_TDM
        // current tile's DMA has landed (TENSORcnt in-order, one outstanding)
        if (wave == 0) __builtin_amdgcn_s_wait_tensorcnt(0);
#else
        // ---------- stage B sync (transposed, f16) ----------
        if (k0 + 64 <= K) {
            float tb[32];
            {
                const float* p = bPtr;
#pragma unroll
                for (int i = 0; i < 32; ++i) { tb[i] = *p; p += ldb; }
            }
            __builtin_prefetch(bPtr + bStep, 0, 1);
#pragma unroll
            for (int c = 0; c < 4; ++c) {
                v8h hb;
#pragma unroll
                for (int i = 0; i < 8; ++i) hb[i] = (_Float16)tb[c * 8 + i];
                *(v8h*)&Bs[sb_col * kLdsStride + sb_k0 + c * 8] = hb;
            }
        } else {
#pragma unroll
            for (int c = 0; c < 4; ++c) {
                v8h hb;
#pragma unroll
                for (int i = 0; i < 8; ++i) {
                    int kk = k0 + sb_k0 + c * 8 + i;
                    int kc = (kk < K) ? kk : (K - 1);
                    float mk = (kk < K) ? 1.f : 0.f;
                    hb[i] = (_Float16)(Bcol[(long long)kc * ldb] * mk);
                }
                *(v8h*)&Bs[sb_col * kLdsStride + sb_k0 + c * 8] = hb;
            }
        }
        bPtr += bStep;
#endif

        __syncthreads();

#if USE_TDM
        // kick off DMA of the NEXT B tile; overlaps with compute below
        if (wave == 0 && (k0 + 64) < K)
            tdm_load_b(Btile + (long long)(k0 + 64) * ldb, ldsWrite,
                       (unsigned)(Ncols - n0), (unsigned)(K - (k0 + 64)), ldb);
#endif

        // ---- 2 WMMAs per staged 64-wide tile ----
#pragma unroll
        for (int s = 0; s < 2; ++s) {
            const _Float16* ap = &As[fa_row * kLdsStride + s * 32 + fa_k];
            v8h a0 = *(const v8h*)(ap);
            v8h a1 = *(const v8h*)(ap + 8);
            v16h afrag, bfrag;
#pragma unroll
            for (int i = 0; i < 8; ++i) { afrag[i] = a0[i]; afrag[i + 8] = a1[i]; }
#if USE_TDM
            const float* bp = bsRead + (s * 32 + fb_k) * 128 + fb_col;
            float fbv[16];
#pragma unroll
            for (int j = 0; j < 16; ++j) fbv[j] = bp[j * 128];
#pragma unroll
            for (int j = 0; j < 16; ++j) bfrag[j] = (_Float16)fbv[j];
#else
            const _Float16* bp2 = &Bs[fb_col * kLdsStride + s * 32 + fb_k];
            v8h b0 = *(const v8h*)(bp2);
            v8h b1 = *(const v8h*)(bp2 + 8);
#pragma unroll
            for (int i = 0; i < 8; ++i) { bfrag[i] = b0[i]; bfrag[i + 8] = b1[i]; }
#endif
            acc = __builtin_amdgcn_wmma_f32_16x16x32_f16(
                false, afrag, false, bfrag, (short)0, acc, false, false);
        }

#if USE_TDM
        // swap double buffers
        const float* tp = bsRead; bsRead = bsWrite; bsWrite = tp;
        unsigned tu = ldsRead; ldsRead = ldsWrite; ldsWrite = tu;
#endif
    }

    // C/D layout: VGPR r -> row m0 + r + (lane>>4)*8, col n0 + wave*16 + (lane&15)
    const int crow0 = m0 + (lane >> 4) * 8;
    const int ccol  = n0 + wave * 16 + (lane & 15);
#pragma unroll
    for (int r = 0; r < 8; ++r) {
        int cr = crow0 + r;
        if (cr < M) Cmat[(long long)cr * ldc + ccol] = acc[r];
    }
}

// =====================================================================
// Graph preparation kernels
// =====================================================================

// At[b][r][j][i] = A[b,r,i,j];  r0 = punct&!aug, r1 = aug   (edge i -> j)
__global__ void build_At(const int* __restrict__ punct, const int* __restrict__ aug,
                         float* __restrict__ At)
{
    int idx = blockIdx.x * blockDim.x + threadIdx.x;   // over [b,i,j]
    if (idx >= kBS * kN * kN) return;
    int j = idx % kN;
    int t = idx / kN;
    int i = t % kN;
    int b = t / kN;
    float a = (float)aug[idx];
    float p = (float)punct[idx];
    long long base = (long long)b * 2 * kN * kN;
    At[base + (long long)j * kN + i]                      = p * (1.f - a);
    At[base + (long long)kN * kN + (long long)j * kN + i] = a;
}

// inv_deg[b,r,j] = 1/deg  (deg = row sum of At[b][r][j][*])
__global__ void calc_inv_deg(const float* __restrict__ At, float* __restrict__ inv_deg)
{
    int idx = blockIdx.x * blockDim.x + threadIdx.x;   // over b*2*N
    if (idx >= kBS * 2 * kN) return;
    const float* row = At + (long long)idx * kN;
    float s = 0.f;
    for (int i = 0; i < kN; ++i) s += row[i];
    inv_deg[idx] = (s > 0.f) ? 1.f / s : 0.f;
}

__global__ void calc_inv_D(const int* __restrict__ s2w, float* __restrict__ inv_D)
{
    int idx = blockIdx.x * blockDim.x + threadIdx.x;   // over (b,n)
    if (idx >= kBS * kN) return;
    const int* row = s2w + (long long)idx * kE;
    float s = 1.f;                                     // global hyperedge always incident
    for (int e = 0; e < kE; ++e) s += (float)row[e];
    inv_D[idx] = 1.f / s;
}

__global__ void calc_inv_B(const int* __restrict__ s2w, float* __restrict__ inv_B)
{
    int idx = blockIdx.x * blockDim.x + threadIdx.x;   // over (b,m)
    if (idx >= kBS * kM) return;
    int m = idx % kM;
    int b = idx / kM;
    if (m == 0) { inv_B[idx] = 1.f / (float)kN; return; }
    float s = 0.f;
    for (int n = 0; n < kN; ++n) s += (float)s2w[((long long)(b * kN + n)) * kE + (m - 1)];
    inv_B[idx] = (s > 0.f) ? 1.f / s : 0.f;
}

// e_attr[b,0,:] = pooled[b];  e_attr[b,1+e,:] = SVO[b,e,:]
__global__ void build_eattr(const float* __restrict__ pooled, const float* __restrict__ svo,
                            float* __restrict__ e_attr)
{
    int idx = blockIdx.x * blockDim.x + threadIdx.x;   // over (b,m,c)
    if (idx >= kBS * kM * kC) return;
    int c = idx % kC;
    int t = idx / kC;
    int m = t % kM;
    int b = t / kM;
    e_attr[idx] = (m == 0) ? pooled[(long long)b * kC + c]
                           : svo[((long long)(b * kE + (m - 1))) * kC + c];
}

__global__ void copy_f32(float* __restrict__ dst, const float* __restrict__ src, int n)
{
    int idx = blockIdx.x * blockDim.x + threadIdx.x;
    if (idx < n) dst[idx] = src[idx];
}

// =====================================================================
// Attention kernels
// =====================================================================

// out[b,n,h] = dot(feat[b,n,h,:], attw[h,:])   feat layout [b,rows,h*C]
__global__ void att_scores(const float* __restrict__ feat, const float* __restrict__ attw,
                           float* __restrict__ out, int rows)
{
    int idx = blockIdx.x * blockDim.x + threadIdx.x;   // over (b,row,h)
    if (idx >= kBS * rows * kH) return;
    int h = idx % kH;
    int t = idx / kH;
    int n = t % rows;
    int b = t / rows;
    const float* x = feat + ((long long)(b * rows + n) * kH + h) * kC;
    const float* w = attw + (long long)h * kC;
    float s = 0.f;
    for (int c = 0; c < kC; ++c) s += x[c] * w[c];
    out[idx] = s;
}

// masked leaky-relu softmax over incident hyperedges; writes alpha in two layouts:
// alphaN[b,h,n,m] and alphaT[b,h,m,n]
__global__ void alpha_softmax(const float* __restrict__ a_n, const float* __restrict__ a_e,
                              const int* __restrict__ s2w,
                              float* __restrict__ alphaN, float* __restrict__ alphaT)
{
    int idx = blockIdx.x * blockDim.x + threadIdx.x;   // over (b,h,n)
    if (idx >= kBS * kH * kN) return;
    int n = idx % kN;
    int t = idx / kN;
    int h = t % kH;
    int b = t / kH;
    float an = a_n[(long long)(b * kN + n) * kH + h];
    const int* inc = s2w + (long long)(b * kN + n) * kE;

    float mx = -3.0e38f;
    for (int m = 0; m < kM; ++m) {
        if (m != 0 && inc[m - 1] == 0) continue;
        float v = an + a_e[(long long)(b * kM + m) * kH + h];
        v = (v > 0.f) ? v : 0.2f * v;
        mx = fmaxf(mx, v);
    }
    float s = 0.f;
    for (int m = 0; m < kM; ++m) {
        if (m != 0 && inc[m - 1] == 0) continue;
        float v = an + a_e[(long long)(b * kM + m) * kH + h];
        v = (v > 0.f) ? v : 0.2f * v;
        s += expf(v - mx);
    }
    float invs = (s > 0.f) ? 1.f / s : 0.f;
    long long baseN = ((long long)(b * kH + h) * kN + n) * kM;
    long long baseT = ((long long)(b * kH + h) * kM) * kN + n;
    for (int m = 0; m < kM; ++m) {
        float p = 0.f;
        if (m == 0 || inc[m - 1] != 0) {
            float v = an + a_e[(long long)(b * kM + m) * kH + h];
            v = (v > 0.f) ? v : 0.2f * v;
            p = expf(v - mx) * invs;
        }
        alphaN[baseN + m]                 = p;
        alphaT[baseT + (long long)m * kN] = p;
    }
}

// msg_e[b,h,m,c] *= inv_B[b,m]
__global__ void scale_msg(float* __restrict__ msg, const float* __restrict__ inv_B)
{
    int idx = blockIdx.x * blockDim.x + threadIdx.x;   // over (b,h,m,c)
    if (idx >= kBS * kH * kM * kC) return;
    int t = idx / kC;
    int m = t % kM;
    int b = t / (kM * kH);
    msg[idx] *= inv_B[b * kM + m];
}

// out_h[b,n,c] = relu( 0.25 * inv_D[b,n] * sum_h outHd[b,h,n,c] + bias[c] )
__global__ void hgcn_epilogue(const float* __restrict__ outHd, const float* __restrict__ inv_D,
                              const float* __restrict__ bias, float* __restrict__ out)
{
    int idx = blockIdx.x * blockDim.x + threadIdx.x;   // over (b,n,c)
    if (idx >= kBS * kN * kC) return;
    int c = idx % kC;
    int t = idx / kC;
    int n = t % kN;
    int b = t / kN;
    float s = 0.f;
#pragma unroll
    for (int h = 0; h < kH; ++h)
        s += outHd[((long long)(b * kH + h) * kN + n) * kC + c];
    float v = 0.25f * inv_D[b * kN + n] * s + bias[c];
    out[idx] = (v > 0.f) ? v : 0.f;
}

// out_r[b,n,c] = relu( agg0*invdeg0 + agg1*invdeg1 + root + bias )
__global__ void rgcn_epilogue(const float* __restrict__ agg0, const float* __restrict__ agg1,
                              const float* __restrict__ root, const float* __restrict__ inv_deg,
                              const float* __restrict__ bias, float* __restrict__ out)
{
    int idx = blockIdx.x * blockDim.x + threadIdx.x;   // over (b,n,c)
    if (idx >= kBS * kN * kC) return;
    int c = idx % kC;
    int t = idx / kC;
    int n = t % kN;
    int b = t / kN;
    float d0 = inv_deg[(b * 2 + 0) * kN + n];
    float d1 = inv_deg[(b * 2 + 1) * kN + n];
    float v = agg0[idx] * d0 + agg1[idx] * d1 + root[idx] + bias[c];
    out[idx] = (v > 0.f) ? v : 0.f;
}

// =====================================================================
// Info-exchange kernels (row 0 of out_r / out_h)
// =====================================================================
__global__ void ie_build(const float* __restrict__ out_r, const float* __restrict__ out_h,
                         float* __restrict__ ctx)
{
    int idx = blockIdx.x * blockDim.x + threadIdx.x;   // over (b, 2C)
    if (idx >= kBS * 2 * kC) return;
    int c = idx % (2 * kC);
    int b = idx / (2 * kC);
    long long row0 = (long long)b * kN * kC;           // node n = 0
    ctx[idx] = (c < kC) ? out_r[row0 + c] : out_h[row0 + (c - kC)];
}

__global__ void ie_relu_bias(float* __restrict__ h1, const float* __restrict__ b1)
{
    int idx = blockIdx.x * blockDim.x + threadIdx.x;   // over (b, 2C)
    if (idx >= kBS * 2 * kC) return;
    float v = h1[idx] + b1[idx % (2 * kC)];
    h1[idx] = (v > 0.f) ? v : 0.f;
}

__global__ void ie_scatter(const float* __restrict__ ctx2, const float* __restrict__ b2,
                           float* __restrict__ out_r, float* __restrict__ out_h)
{
    int idx = blockIdx.x * blockDim.x + threadIdx.x;   // over (b, 2C)
    if (idx >= kBS * 2 * kC) return;
    int c = idx % (2 * kC);
    int b = idx / (2 * kC);
    float v = ctx2[idx] + b2[c];
    long long row0 = (long long)b * kN * kC;
    if (c < kC) out_r[row0 + c] = v;
    else        out_h[row0 + (c - kC)] = v;
}

__global__ void write_out(const float* __restrict__ xr, const float* __restrict__ xh,
                          float* __restrict__ out)
{
    int idx = blockIdx.x * blockDim.x + threadIdx.x;
    const int half = kBS * kN * kC;
    if (idx >= 2 * half) return;
    out[idx] = (idx < half) ? xr[idx] : xh[idx - half];
}

// =====================================================================
// Host orchestration
// =====================================================================
static inline void launch_gemm(const float* A, const float* B, float* C,
                               int M, int Ncols, int K,
                               int lda, int ldb, int ldc,
                               long long sA, long long sB, long long sC,
                               int batch, hipStream_t s)
{
    dim3 grid(CDIV(Ncols, 128), CDIV(M, 16), batch);
    wmma_gemm_f16<<<grid, 256, 0, s>>>(A, B, C, M, Ncols, K, lda, ldb, ldc, sA, sB, sC);
}

extern "C" void kernel_launch(void* const* d_in, const int* in_sizes, int n_in,
                              void* d_out, int out_size, void* d_ws, size_t ws_size,
                              hipStream_t stream)
{
    const float* encoded = (const float*)d_in[0];
    const float* svo     = (const float*)d_in[1];
    const float* pooled  = (const float*)d_in[2];
    const int*   s2w     = (const int*)d_in[3];
    const int*   aug     = (const int*)d_in[4];
    const int*   punct   = (const int*)d_in[5];
    const float* w_rel   = (const float*)d_in[6];
    const float* w_root  = (const float*)d_in[7];
    const float* b_rgcn  = (const float*)d_in[8];
    const float* w_lin   = (const float*)d_in[9];
    const float* att_x   = (const float*)d_in[10];
    const float* att_e   = (const float*)d_in[11];
    const float* b_hgcn  = (const float*)d_in[12];
    const float* ie_w1   = (const float*)d_in[13];
    const float* ie_b1   = (const float*)d_in[14];
    const float* ie_w2   = (const float*)d_in[15];
    const float* ie_b2   = (const float*)d_in[16];

    float* W = (float*)d_ws;
    size_t off = 0;
    auto alloc = [&](size_t n) { float* p = W + off; off += n; return p; };

    float* At      = alloc((size_t)kBS * 2 * kN * kN);
    float* inv_deg = alloc((size_t)kBS * 2 * kN);
    float* inv_D   = alloc((size_t)kBS * kN);
    float* inv_B   = alloc((size_t)kBS * kM);
    float* e_attr  = alloc((size_t)kBS * kM * kC);
    float* xr0     = alloc((size_t)kBS * kN * kC);
    float* xh0     = alloc((size_t)kBS * kN * kC);
    float* xr1     = alloc((size_t)kBS * kN * kC);
    float* xh1     = alloc((size_t)kBS * kN * kC);
    float* xl      = alloc((size_t)kBS * kN * kH * kC);
    float* el      = alloc((size_t)kBS * kM * kH * kC);
    float* a_n     = alloc((size_t)kBS * kN * kH);
    float* a_e     = alloc((size_t)kBS * kM * kH);
    float* alphaN  = alloc((size_t)kBS * kH * kN * kM);
    float* alphaT  = alloc((size_t)kBS * kH * kM * kN);
    float* msg_e   = alloc((size_t)kBS * kH * kM * kC);
    float* outHd   = alloc((size_t)kBS * kH * kN * kC);
    float* xw0     = alloc((size_t)kBS * kN * kC);
    float* xw1     = alloc((size_t)kBS * kN * kC);
    float* root    = alloc((size_t)kBS * kN * kC);
    float* agg0    = alloc((size_t)kBS * kN * kC);
    float* agg1    = alloc((size_t)kBS * kN * kC);
    float* ctx     = alloc((size_t)kBS * 2 * kC);
    float* h1      = alloc((size_t)kBS * 2 * kC);
    float* ctx2    = alloc((size_t)kBS * 2 * kC);

    const int TB = 256;
    const int nodesFeat = kBS * kN * kC;

    // ---- graph prep ----
    build_At<<<CDIV(kBS * kN * kN, TB), TB, 0, stream>>>(punct, aug, At);
    calc_inv_deg<<<CDIV(kBS * 2 * kN, TB), TB, 0, stream>>>(At, inv_deg);
    calc_inv_D<<<CDIV(kBS * kN, TB), TB, 0, stream>>>(s2w, inv_D);
    calc_inv_B<<<CDIV(kBS * kM, TB), TB, 0, stream>>>(s2w, inv_B);
    build_eattr<<<CDIV(kBS * kM * kC, TB), TB, 0, stream>>>(pooled, svo, e_attr);
    copy_f32<<<CDIV(nodesFeat, TB), TB, 0, stream>>>(xr0, encoded, nodesFeat);
    copy_f32<<<CDIV(nodesFeat, TB), TB, 0, stream>>>(xh0, encoded, nodesFeat);

    float* xr_cur = xr0; float* xh_cur = xh0;
    float* xr_nxt = xr1; float* xh_nxt = xh1;

    for (int l = 0; l < kL; ++l) {
        const float* wl  = w_lin + (long long)l * kC * kH * kC;
        const float* axw = att_x + (long long)l * kH * kC;
        const float* aew = att_e + (long long)l * kH * kC;
        const float* bh  = b_hgcn + (long long)l * kC;
        const float* br  = b_rgcn + (long long)l * kC;
        const float* wrt = w_root + (long long)l * kC * kC;
        const float* w1  = ie_w1 + (long long)l * 2 * kC * 2 * kC;
        const float* b1  = ie_b1 + (long long)l * 2 * kC;
        const float* w2  = ie_w2 + (long long)l * 2 * kC * 2 * kC;
        const float* b2  = ie_b2 + (long long)l * 2 * kC;

        // ---- HypergraphConv ----
        launch_gemm(xh_cur, wl, xl, kN, kH * kC, kC,
                    kC, kH * kC, kH * kC,
                    (long long)kN * kC, 0, (long long)kN * kH * kC, kBS, stream);
        launch_gemm(e_attr, wl, el, kM, kH * kC, kC,
                    kC, kH * kC, kH * kC,
                    (long long)kM * kC, 0, (long long)kM * kH * kC, kBS, stream);

        att_scores<<<CDIV(kBS * kN * kH, TB), TB, 0, stream>>>(xl, axw, a_n, kN);
        att_scores<<<CDIV(kBS * kM * kH, TB), TB, 0, stream>>>(el, aew, a_e, kM);
        alpha_softmax<<<CDIV(kBS * kH * kN, TB), TB, 0, stream>>>(a_n, a_e, s2w, alphaN, alphaT);

        // msg_e[b,h] = alphaT[b,h] (65x256) @ xl[b,:,h,:] (256x512)
        for (int h = 0; h < kH; ++h) {
            launch_gemm(alphaT + (long long)h * kM * kN,
                        xl + (long long)h * kC,
                        msg_e + (long long)h * kM * kC,
                        kM, kC, kN,
                        kN, kH * kC, kC,
                        (long long)kH * kM * kN, (long long)kN * kH * kC, (long long)kH * kM * kC,
                        kBS, stream);
        }
        scale_msg<<<CDIV(kBS * kH * kM * kC, TB), TB, 0, stream>>>(msg_e, inv_B);
        // outHd[b,h] = alphaN[b,h] (256x65) @ msg_e[b,h] (65x512)
        for (int h = 0; h < kH; ++h) {
            launch_gemm(alphaN + (long long)h * kN * kM,
                        msg_e + (long long)h * kM * kC,
                        outHd + (long long)h * kN * kC,
                        kN, kC, kM,
                        kM, kC, kC,
                        (long long)kH * kN * kM, (long long)kH * kM * kC, (long long)kH * kN * kC,
                        kBS, stream);
        }
        hgcn_epilogue<<<CDIV(nodesFeat, TB), TB, 0, stream>>>(outHd, inv_D, bh, xh_nxt);

        // ---- RGCN ----
        for (int r = 0; r < 2; ++r) {
            const float* wr = w_rel + (long long)(l * 2 + r) * kC * kC;
            float* xw = (r == 0) ? xw0 : xw1;
            launch_gemm(xr_cur, wr, xw, kN, kC, kC,
                        kC, kC, kC,
                        (long long)kN * kC, 0, (long long)kN * kC, kBS, stream);
        }
        launch_gemm(xr_cur, wrt, root, kN, kC, kC,
                    kC, kC, kC,
                    (long long)kN * kC, 0, (long long)kN * kC, kBS, stream);
        for (int r = 0; r < 2; ++r) {
            float* xw  = (r == 0) ? xw0 : xw1;
            float* agg = (r == 0) ? agg0 : agg1;
            launch_gemm(At + (long long)r * kN * kN, xw, agg,
                        kN, kC, kN,
                        kN, kC, kC,
                        2LL * kN * kN, (long long)kN * kC, (long long)kN * kC, kBS, stream);
        }
        rgcn_epilogue<<<CDIV(nodesFeat, TB), TB, 0, stream>>>(agg0, agg1, root, inv_deg, br, xr_nxt);

        // ---- info exchange (node 0) ----
        ie_build<<<CDIV(kBS * 2 * kC, TB), TB, 0, stream>>>(xr_nxt, xh_nxt, ctx);
        launch_gemm(ctx, w1, h1, kBS, 2 * kC, 2 * kC,
                    2 * kC, 2 * kC, 2 * kC, 0, 0, 0, 1, stream);
        ie_relu_bias<<<CDIV(kBS * 2 * kC, TB), TB, 0, stream>>>(h1, b1);
        launch_gemm(h1, w2, ctx2, kBS, 2 * kC, 2 * kC,
                    2 * kC, 2 * kC, 2 * kC, 0, 0, 0, 1, stream);
        ie_scatter<<<CDIV(kBS * 2 * kC, TB), TB, 0, stream>>>(ctx2, b2, xr_nxt, xh_nxt);

        // ping-pong
        float* t;
        t = xr_cur; xr_cur = xr_nxt; xr_nxt = t;
        t = xh_cur; xh_cur = xh_nxt; xh_nxt = t;
    }

    write_out<<<CDIV(2 * nodesFeat, TB), TB, 0, stream>>>(xr_cur, xh_cur, (float*)d_out);

    (void)in_sizes; (void)n_in; (void)out_size; (void)ws_size;
}
